// LongTermMemory_46901042872833
// MI455X (gfx1250) — matrix operand order repaired and checked by reading, not compiled, not run on hardware
//
#include <hip/hip_runtime.h>
#include <hip/hip_bf16.h>
#include <math.h>

typedef __attribute__((ext_vector_type(16))) __bf16 v16bf;
typedef __attribute__((ext_vector_type(8)))  float  v8f;
typedef unsigned short u16;

#define TILE_K 32
#define LDT    40   // padded LDS row stride in elements (80B rows, 16B aligned)

#if __has_builtin(__builtin_amdgcn_global_load_async_to_lds_b128)
#define HAVE_ASYNC_LDS 1
typedef int v4i_ __attribute__((vector_size(16)));
typedef v4i_ __attribute__((address_space(1))) *gv4ip;   // global (AS1) V4i*
typedef v4i_ __attribute__((address_space(3))) *lv4ip;   // LDS (AS3) V4i*
#endif

static __device__ __forceinline__ void wait_async0() {
#if defined(HAVE_ASYNC_LDS)
#if __has_builtin(__builtin_amdgcn_s_wait_asynccnt)
  __builtin_amdgcn_s_wait_asynccnt(0);
#else
  asm volatile("s_wait_asynccnt 0x0" ::: "memory");
#endif
#endif
}

union Frag16 {
  v16bf v;
  uint4 q[2];
};

static __device__ __forceinline__ __bf16 f2bf(float x) { return (__bf16)x; }

// ---------------------------------------------------------------------------
// fp32 -> bf16 elementwise
// ---------------------------------------------------------------------------
__global__ __launch_bounds__(256) void f32_to_bf16_kernel(const float* __restrict__ in,
                                                          __bf16* __restrict__ out,
                                                          size_t n) {
  size_t i = (size_t)blockIdx.x * 256 + threadIdx.x;
  if (i < n) out[i] = f2bf(in[i]);
}

// ---------------------------------------------------------------------------
// W (K x N, fp32, row-major) -> Wt (N x K, bf16). Coalesced writes over out.
// ---------------------------------------------------------------------------
__global__ __launch_bounds__(256) void transpose_bf16_kernel(const float* __restrict__ in,
                                                             __bf16* __restrict__ out,
                                                             int K, int N) {
  size_t i = (size_t)blockIdx.x * 256 + threadIdx.x;
  size_t total = (size_t)K * N;
  if (i < total) {
    int n = (int)(i / K);
    int k = (int)(i % K);
    out[(size_t)n * K + k] = f2bf(in[(size_t)k * N + n]);
  }
}

// ---------------------------------------------------------------------------
// WMMA bf16 GEMM: C(MxN) = A(MxK) @ Bt(NxK)^T  with epilogues
//   EPI 0: Cf = acc                         (fp32 out)
//   EPI 1: Cb = bf16(relu(acc + bias[n]))   (MLP layer 1)
//   EPI 3: Cf = acc + bias[n] - aux[m,n]    (signal = mlp(xk) - xv)
//   EPI 4: Cb = bf16(acc + bias[n])         (MLP layer 2, q path)
// Block tile 128x128, K-step 32. 8 waves (wave32): 4x2 grid, each 32x64.
// Async global->LDS double-buffered pipeline when available.
// ---------------------------------------------------------------------------
template<int EPI>
__global__ __launch_bounds__(256) void gemm_bf16_wmma(
    const __bf16* __restrict__ A, const __bf16* __restrict__ Bt,
    float* __restrict__ Cf, __bf16* __restrict__ Cb,
    const float* __restrict__ bias, const float* __restrict__ aux,
    int M, int N, int K)
{
#if defined(HAVE_ASYNC_LDS)
  __shared__ __align__(16) u16 Al[2][128 * LDT];
  __shared__ __align__(16) u16 Bl[2][128 * LDT];
#else
  __shared__ __align__(16) u16 Al[1][128 * LDT];
  __shared__ __align__(16) u16 Bl[1][128 * LDT];
#endif

  const int tid  = threadIdx.x;
  const int lane = tid & 31;
  const int wave = tid >> 5;
  const int half = lane >> 4;   // which 16-lane half
  const int lrow = lane & 15;
  const int wm   = wave >> 1;   // 0..3 -> M offset wm*32
  const int wn   = wave & 1;    // 0..1 -> N offset wn*64
  const int row0 = blockIdx.y * 128;
  const int col0 = blockIdx.x * 128;

  const u16* Au = (const u16*)A;
  const u16* Bu = (const u16*)Bt;

  v8f acc[2][4];
  for (int i = 0; i < 2; ++i)
    for (int j = 0; j < 4; ++j)
      for (int e = 0; e < 8; ++e) acc[i][j][e] = 0.0f;

#if defined(HAVE_ASYNC_LDS)
  // Issue one 128x32 bf16 tile (A and B) as async DMA into LDS buffer `buf`.
  auto load_tile_async = [&](int k0, int buf) {
    for (int i = tid; i < 512; i += 256) {
      int r = i >> 2;
      int c = (i & 3) << 3;  // element offset, 8 x u16 = 16B
      const u16* ga = &Au[(size_t)(row0 + r) * K + k0 + c];
      const u16* gb = &Bu[(size_t)(col0 + r) * K + k0 + c];
      u16* la = &Al[buf][r * LDT + c];
      u16* lb = &Bl[buf][r * LDT + c];
      __builtin_amdgcn_global_load_async_to_lds_b128(
          (gv4ip)(unsigned long long)ga,
          (lv4ip)(unsigned int)(unsigned long long)la, 0, 0);
      __builtin_amdgcn_global_load_async_to_lds_b128(
          (gv4ip)(unsigned long long)gb,
          (lv4ip)(unsigned int)(unsigned long long)lb, 0, 0);
    }
  };

  load_tile_async(0, 0);
  wait_async0();
  __syncthreads();

  int it = 0;
  for (int k0 = 0; k0 < K; k0 += TILE_K, ++it) {
    const int cur = it & 1;
    const bool more = (k0 + TILE_K) < K;
    if (more) load_tile_async(k0 + TILE_K, cur ^ 1);

    // A fragment (16x32): lane holds row m=lrow, K = half*8 + {0..7} and {16..23}
    Frag16 af[2];
    for (int mt = 0; mt < 2; ++mt) {
      const u16* p = &Al[cur][(wm * 32 + mt * 16 + lrow) * LDT + half * 8];
      af[mt].q[0] = *(const uint4*)p;
      af[mt].q[1] = *(const uint4*)(p + 16);
    }
    // B fragment (32x16): lane holds col n=lrow, K = half*8 + {0..7} and {16..23}
    Frag16 bfrag[4];
    for (int nt = 0; nt < 4; ++nt) {
      const u16* p = &Bl[cur][(wn * 64 + nt * 16 + lrow) * LDT + half * 8];
      bfrag[nt].q[0] = *(const uint4*)p;
      bfrag[nt].q[1] = *(const uint4*)(p + 16);
    }

    for (int mt = 0; mt < 2; ++mt)
      for (int nt = 0; nt < 4; ++nt)
        acc[mt][nt] = __builtin_amdgcn_wmma_f32_16x16x32_bf16(
            false, af[mt].v, false, bfrag[nt].v, (short)0, acc[mt][nt],
            false, false);

    if (more) wait_async0();
    __syncthreads();
  }
#else
  for (int k0 = 0; k0 < K; k0 += TILE_K) {
    for (int i = tid; i < 512; i += 256) {
      int r = i >> 2;
      int c = (i & 3) << 3;
      *(uint4*)&Al[0][r * LDT + c] = *(const uint4*)&Au[(size_t)(row0 + r) * K + k0 + c];
      *(uint4*)&Bl[0][r * LDT + c] = *(const uint4*)&Bu[(size_t)(col0 + r) * K + k0 + c];
      if (k0 + TILE_K < K) {
        __builtin_prefetch(&Au[(size_t)(row0 + r) * K + k0 + TILE_K + c], 0, 1);
        __builtin_prefetch(&Bu[(size_t)(col0 + r) * K + k0 + TILE_K + c], 0, 1);
      }
    }
    __syncthreads();

    Frag16 af[2];
    for (int mt = 0; mt < 2; ++mt) {
      const u16* p = &Al[0][(wm * 32 + mt * 16 + lrow) * LDT + half * 8];
      af[mt].q[0] = *(const uint4*)p;
      af[mt].q[1] = *(const uint4*)(p + 16);
    }
    Frag16 bfrag[4];
    for (int nt = 0; nt < 4; ++nt) {
      const u16* p = &Bl[0][(wn * 64 + nt * 16 + lrow) * LDT + half * 8];
      bfrag[nt].q[0] = *(const uint4*)p;
      bfrag[nt].q[1] = *(const uint4*)(p + 16);
    }

    for (int mt = 0; mt < 2; ++mt)
      for (int nt = 0; nt < 4; ++nt)
        acc[mt][nt] = __builtin_amdgcn_wmma_f32_16x16x32_bf16(
            false, af[mt].v, false, bfrag[nt].v, (short)0, acc[mt][nt],
            false, false);
    __syncthreads();
  }
#endif

  // Epilogue. C layout: VGPR v -> M = half*8 + v, lane lrow -> N.
  for (int mt = 0; mt < 2; ++mt)
    for (int nt = 0; nt < 4; ++nt) {
      int n = col0 + wn * 64 + nt * 16 + lrow;
      float bv = (EPI != 0) ? bias[n] : 0.0f;
      for (int v = 0; v < 8; ++v) {
        int m = row0 + wm * 32 + mt * 16 + half * 8 + v;
        size_t idx = (size_t)m * N + n;
        float val = acc[mt][nt][v];
        if (EPI == 0) {
          Cf[idx] = val;
        } else if (EPI == 1) {
          val += bv; val = val > 0.0f ? val : 0.0f;
          Cb[idx] = f2bf(val);
        } else if (EPI == 3) {
          Cf[idx] = val + bv - aux[idx];
        } else if (EPI == 4) {
          Cb[idx] = f2bf(val + bv);
        }
      }
    }
}

// ---------------------------------------------------------------------------
// Fused LayerNorm + RoPE: one block per row (B*S rows of D). Output bf16.
// ---------------------------------------------------------------------------
__global__ __launch_bounds__(256) void ln_rope_kernel(
    const float* __restrict__ Y, const float* __restrict__ w,
    const float* __restrict__ bparm, const float* __restrict__ cosv,
    const float* __restrict__ sinv, __bf16* __restrict__ out,
    int S, int D, int halfHD)
{
  __shared__ float red[256];
  const int r = blockIdx.x;
  const int s = r % S;
  const float* y = Y + (size_t)r * D;

  float sum = 0.0f, ssq = 0.0f;
  for (int d = threadIdx.x; d < D; d += 256) {
    float v = y[d];
    sum += v; ssq += v * v;
  }
  red[threadIdx.x] = sum; __syncthreads();
  for (int o = 128; o > 0; o >>= 1) {
    if (threadIdx.x < o) red[threadIdx.x] += red[threadIdx.x + o];
    __syncthreads();
  }
  float mu = red[0] / (float)D;
  __syncthreads();
  red[threadIdx.x] = ssq; __syncthreads();
  for (int o = 128; o > 0; o >>= 1) {
    if (threadIdx.x < o) red[threadIdx.x] += red[threadIdx.x + o];
    __syncthreads();
  }
  float var = red[0] / (float)D - mu * mu;
  float rinv = rsqrtf(var + 1e-5f);

  const int pairs = D >> 1;
  for (int p = threadIdx.x; p < pairs; p += 256) {
    int d0 = p << 1;
    float z0 = (y[d0]     - mu) * rinv * w[d0]     + bparm[d0];
    float z1 = (y[d0 + 1] - mu) * rinv * w[d0 + 1] + bparm[d0 + 1];
    int i = p & (halfHD - 1);   // pair index inside head (halfHD = 64, pow2)
    float c  = cosv[(size_t)s * halfHD + i];
    float sn = sinv[(size_t)s * halfHD + i];
    out[(size_t)r * D + d0]     = f2bf(z0 * c - z1 * sn);
    out[(size_t)r * D + d0 + 1] = f2bf(z0 * sn + z1 * c);
  }
}

// ---------------------------------------------------------------------------
// Gates: alpha = sigmoid(x@a_w+b), eta = sigmoid(x@e_w+b), theta = relu(x@t_w+b)
// ---------------------------------------------------------------------------
__global__ __launch_bounds__(64) void gates_kernel(
    const float* __restrict__ x,
    const float* __restrict__ aw, const float* __restrict__ ab,
    const float* __restrict__ ew, const float* __restrict__ eb,
    const float* __restrict__ tw, const float* __restrict__ tb,
    float* __restrict__ alpha, float* __restrict__ eta, float* __restrict__ theta,
    int D, int H)
{
  const int r = blockIdx.x;
  const int t = threadIdx.x;
  if (t >= 3 * H) return;
  const int which = t / H;
  const int j = t % H;
  const float* wsel = (which == 0) ? aw : (which == 1) ? ew : tw;
  const float* xr = x + (size_t)r * D;
  float acc = 0.0f;
  for (int d = 0; d < D; ++d) acc += xr[d] * wsel[(size_t)d * H + j];
  if (which == 0) {
    acc += ab[j];
    alpha[(size_t)r * H + j] = 1.0f / (1.0f + expf(-acc));
  } else if (which == 1) {
    acc += eb[j];
    eta[(size_t)r * H + j] = 1.0f / (1.0f + expf(-acc));
  } else {
    acc += tb[j];
    theta[(size_t)r * H + j] = acc > 0.0f ? acc : 0.0f;
  }
}

// ---------------------------------------------------------------------------
// Sequential scan over S; one block per (b,h), one thread per hd element.
// ---------------------------------------------------------------------------
__global__ __launch_bounds__(128) void scan_kernel(
    const float* __restrict__ eta, const float* __restrict__ theta,
    const float* __restrict__ alpha, const float* __restrict__ signal,
    const float* __restrict__ mem0, const float* __restrict__ sur0,
    float* __restrict__ mem_out, float* __restrict__ sur_out,
    int Bc, int S, int H, int HD)
{
  const int b = blockIdx.x / H;
  const int h = blockIdx.x % H;
  const int t = threadIdx.x;
  float sur = sur0[(size_t)h * HD + t];
  float mem = mem0[(size_t)h * HD + t];
  for (int s = 0; s < S; ++s) {
    size_t gi = ((size_t)b * S + s) * H + h;
    float e  = eta[gi];
    float th = theta[gi];
    float al = alpha[gi];
    float sg = signal[gi * HD + t];
    sur = e * sur - th * sg;
    mem = (1.0f - al) * mem + sur;
  }
  mem_out[((size_t)b * H + h) * HD + t] = mem;
  sur_out[((size_t)b * H + h) * HD + t] = sur;
}

// ---------------------------------------------------------------------------
// Host orchestration
// ---------------------------------------------------------------------------
extern "C" void kernel_launch(void* const* d_in, const int* in_sizes, int n_in,
                              void* d_out, int out_size, void* d_ws, size_t ws_size,
                              hipStream_t stream) {
  const int Bc = 4, S = 2048, D = 2048, H = 16, HD = 128;
  const size_t R = (size_t)Bc * S;          // 8192 rows
  (void)in_sizes; (void)n_in; (void)out_size; (void)ws_size;

  const float* x    = (const float*)d_in[0];
  const float* fc   = (const float*)d_in[1];
  const float* fs   = (const float*)d_in[2];
  const float* wq   = (const float*)d_in[3];
  const float* wk   = (const float*)d_in[4];
  const float* wv   = (const float*)d_in[5];
  const float* wo   = (const float*)d_in[6];
  const float* qn_w = (const float*)d_in[7];
  const float* qn_b = (const float*)d_in[8];
  const float* kn_w = (const float*)d_in[9];
  const float* kn_b = (const float*)d_in[10];
  const float* m1_w = (const float*)d_in[11];
  const float* m1_b = (const float*)d_in[12];
  const float* m2_w = (const float*)d_in[13];
  const float* m2_b = (const float*)d_in[14];
  const float* a_w  = (const float*)d_in[15];
  const float* a_b  = (const float*)d_in[16];
  const float* e_w  = (const float*)d_in[17];
  const float* e_b  = (const float*)d_in[18];
  const float* t_w  = (const float*)d_in[19];
  const float* t_b  = (const float*)d_in[20];
  const float* mem0 = (const float*)d_in[21];
  const float* sur0 = (const float*)d_in[22];

  char* ws = (char*)d_ws;
  size_t off = 0;
  auto alloc = [&](size_t bytes) -> char* {
    char* p = ws + off;
    off += (bytes + 255) & ~(size_t)255;
    return p;
  };

  __bf16* XB  = (__bf16*)alloc(R * D * 2);
  __bf16* WQt = (__bf16*)alloc((size_t)D * D * 2);
  __bf16* WKt = (__bf16*)alloc((size_t)D * D * 2);
  __bf16* WVt = (__bf16*)alloc((size_t)D * D * 2);
  __bf16* WOt = (__bf16*)alloc((size_t)D * D * 2);
  __bf16* M1t = (__bf16*)alloc((size_t)HD * HD * 2);
  __bf16* M2t = (__bf16*)alloc((size_t)HD * HD * 2);
  float*  G0  = (float*)alloc(R * D * 4);   // q_raw
  float*  G1  = (float*)alloc(R * D * 4);   // k_raw -> signal
  float*  G2  = (float*)alloc(R * D * 4);   // v
  __bf16* QR  = (__bf16*)alloc(R * D * 2);  // rope(ln(q))
  __bf16* KR  = (__bf16*)alloc(R * D * 2);  // rope(ln(k))
  __bf16* H1  = (__bf16*)alloc(R * D * 2);  // mlp hidden
  __bf16* OQ  = (__bf16*)alloc(R * D * 2);  // mlp(xq) bf16
  float*  ALP = (float*)alloc(R * H * 4);
  float*  ETA = (float*)alloc(R * H * 4);
  float*  THE = (float*)alloc(R * H * 4);

  float* out_ptr = (float*)d_out;
  float* mem_out = out_ptr + R * D;
  float* sur_out = mem_out + (size_t)Bc * H * HD;

  // 1. Convert activations / weights to bf16 (weights transposed to NxK).
  {
    size_t n = R * D;
    f32_to_bf16_kernel<<<dim3((unsigned)((n + 255) / 256)), dim3(256), 0, stream>>>(x, XB, n);
    unsigned gDD = (unsigned)(((size_t)D * D + 255) / 256);
    transpose_bf16_kernel<<<gDD, 256, 0, stream>>>(wq, WQt, D, D);
    transpose_bf16_kernel<<<gDD, 256, 0, stream>>>(wk, WKt, D, D);
    transpose_bf16_kernel<<<gDD, 256, 0, stream>>>(wv, WVt, D, D);
    transpose_bf16_kernel<<<gDD, 256, 0, stream>>>(wo, WOt, D, D);
    unsigned gHH = (unsigned)(((size_t)HD * HD + 255) / 256);
    transpose_bf16_kernel<<<gHH, 256, 0, stream>>>(m1_w, M1t, HD, HD);
    transpose_bf16_kernel<<<gHH, 256, 0, stream>>>(m2_w, M2t, HD, HD);
  }

  // 2. Projections: q_raw, k_raw, v  (M=8192, N=2048, K=2048)
  dim3 gProj(D / 128, (unsigned)(R / 128));
  gemm_bf16_wmma<0><<<gProj, 256, 0, stream>>>(XB, WQt, G0, nullptr, nullptr, nullptr, (int)R, D, D);
  gemm_bf16_wmma<0><<<gProj, 256, 0, stream>>>(XB, WKt, G1, nullptr, nullptr, nullptr, (int)R, D, D);
  gemm_bf16_wmma<0><<<gProj, 256, 0, stream>>>(XB, WVt, G2, nullptr, nullptr, nullptr, (int)R, D, D);

  // 3. LN + RoPE on q and k -> bf16
  ln_rope_kernel<<<(unsigned)R, 256, 0, stream>>>(G0, qn_w, qn_b, fc, fs, QR, S, D, HD / 2);
  ln_rope_kernel<<<(unsigned)R, 256, 0, stream>>>(G1, kn_w, kn_b, fc, fs, KR, S, D, HD / 2);

  // 4. Gates
  gates_kernel<<<(unsigned)R, 64, 0, stream>>>(x, a_w, a_b, e_w, e_b, t_w, t_b,
                                               ALP, ETA, THE, D, H);

  // 5. signal = mlp(KR) - v.  Per-head MLP as (R*H, HD) @ (HD, HD).
  const int Mh = (int)(R * H);
  dim3 gMlp(HD / 128, (unsigned)(Mh / 128));
  gemm_bf16_wmma<1><<<gMlp, 256, 0, stream>>>(KR, M1t, nullptr, H1, m1_b, nullptr, Mh, HD, HD);
  gemm_bf16_wmma<3><<<gMlp, 256, 0, stream>>>(H1, M2t, G1, nullptr, m2_b, G2, Mh, HD, HD);

  // 6. Scan over S -> mem_f, sur_f
  scan_kernel<<<(unsigned)(Bc * H), HD, 0, stream>>>(ETA, THE, ALP, G1, mem0, sur0,
                                                     mem_out, sur_out, Bc, S, H, HD);

  // 7. out = mlp(QR) @ wo
  gemm_bf16_wmma<1><<<gMlp, 256, 0, stream>>>(QR, M1t, nullptr, H1, m1_b, nullptr, Mh, HD, HD);
  gemm_bf16_wmma<4><<<gMlp, 256, 0, stream>>>(H1, M2t, nullptr, OQ, m2_b, nullptr, Mh, HD, HD);
  gemm_bf16_wmma<0><<<gProj, 256, 0, stream>>>(OQ, WOt, out_ptr, nullptr, nullptr, nullptr, (int)R, D, D);
}
